// self_attention_80298708566343
// MI455X (gfx1250) — matrix operand order repaired and checked by reading, not compiled, not run on hardware
//
#include <hip/hip_runtime.h>
#include <hip/hip_bf16.h>
#include <math.h>

// ---------------------------------------------------------------------------
// Types for CDNA5 WMMA (wave32): A/B = 16 bf16 per lane, C/D = 8 f32 per lane
// ---------------------------------------------------------------------------
typedef __attribute__((ext_vector_type(16))) __bf16 v16bf;
typedef __attribute__((ext_vector_type(8)))  __bf16 v8bf;
typedef __attribute__((ext_vector_type(8)))  float  v8f;

#define BB 32
#define SS 512
#define EE 512
#define NL 3
#define NTOK (BB * SS)   // 16384 rows

// ---------------------------------------------------------------------------
// f32 -> bf16 conversion (weights)
// ---------------------------------------------------------------------------
__global__ void f32_to_bf16_k(const float* __restrict__ in, __bf16* __restrict__ out, int n) {
    int i = blockIdx.x * blockDim.x + threadIdx.x;
    if (i < n) out[i] = (__bf16)in[i];
}

// ---------------------------------------------------------------------------
// h = emb[x] + positional encoding   (one block per (b,s) row, 256 threads)
// ---------------------------------------------------------------------------
__global__ void embed_pos_k(const int* __restrict__ x, const float* __restrict__ emb,
                            float* __restrict__ h) {
    int row = blockIdx.x;           // b*S + s
    int s   = row & (SS - 1);
    int tok = x[row];
    const float neg_ln_over_d = -(logf(10000.0f) / (float)EE);
    #pragma unroll
    for (int k = 0; k < 2; ++k) {
        int e   = threadIdx.x + k * 256;
        float dv = __expf(neg_ln_over_d * (float)(e & ~1));
        float ang = (float)s * dv;
        float pe  = (e & 1) ? __cosf(ang) : __sinf(ang);
        h[(size_t)row * EE + e] = emb[(size_t)tok * EE + e] + pe;
    }
}

// ---------------------------------------------------------------------------
// BatchNorm over (B,S) per feature: one block per feature e
// ---------------------------------------------------------------------------
__global__ void bn_stats_k(const float* __restrict__ h, float* __restrict__ mean,
                           float* __restrict__ rstd) {
    __shared__ float s1[256], s2[256];
    int e = blockIdx.x, t = threadIdx.x;
    float s = 0.f, q = 0.f;
    for (int r = t; r < NTOK; r += 256) {
        float v = h[(size_t)r * EE + e];
        s += v; q += v * v;
    }
    s1[t] = s; s2[t] = q;
    __syncthreads();
    for (int st = 128; st > 0; st >>= 1) {
        if (t < st) { s1[t] += s1[t + st]; s2[t] += s2[t + st]; }
        __syncthreads();
    }
    if (t == 0) {
        float m = s1[0] * (1.0f / NTOK);
        float v = s2[0] * (1.0f / NTOK) - m * m;
        mean[e] = m;
        rstd[e] = rsqrtf(v + 1e-5f);
    }
}

__global__ void bn_apply_k(const float* __restrict__ h, const float* __restrict__ mean,
                           const float* __restrict__ rstd, __bf16* __restrict__ hn) {
    size_t i = (size_t)blockIdx.x * blockDim.x + threadIdx.x;
    if (i < (size_t)NTOK * EE) {
        int e = (int)(i & (EE - 1));
        hn[i] = (__bf16)((h[i] - mean[e]) * rstd[e]);
    }
}

// ---------------------------------------------------------------------------
// Row softmax over 512 cols; one wave32 per row; writes bf16 probabilities
// ---------------------------------------------------------------------------
__global__ void softmax_rows_k(const float* __restrict__ sc, __bf16* __restrict__ a) {
    int row  = blockIdx.x * 8 + (threadIdx.x >> 5);
    int lane = threadIdx.x & 31;
    const float* p = sc + (size_t)row * SS;
    float v[16];
    float m = -3.402823e38f;
    #pragma unroll
    for (int i = 0; i < 16; ++i) { v[i] = p[i * 32 + lane]; m = fmaxf(m, v[i]); }
    #pragma unroll
    for (int d = 16; d > 0; d >>= 1) m = fmaxf(m, __shfl_xor(m, d, 32));
    float s = 0.f;
    #pragma unroll
    for (int i = 0; i < 16; ++i) { v[i] = __expf(v[i] - m); s += v[i]; }
    #pragma unroll
    for (int d = 16; d > 0; d >>= 1) s += __shfl_xor(s, d, 32);
    float r = 1.0f / s;
    __bf16* q = a + (size_t)row * SS;
    #pragma unroll
    for (int i = 0; i < 16; ++i) q[i * 32 + lane] = (__bf16)(v[i] * r);
}

// ---------------------------------------------------------------------------
// hbar[b,:] = mean over s of h[b,s,:]  (f32 + bf16 copies)
// ---------------------------------------------------------------------------
__global__ void mean_seq_k(const float* __restrict__ h, float* __restrict__ hbar,
                           __bf16* __restrict__ hbarb) {
    int b = blockIdx.x, t = threadIdx.x;
    for (int e = t; e < EE; e += 256) {
        const float* p = h + (size_t)b * SS * EE + e;
        float s = 0.f;
        for (int s2 = 0; s2 < SS; ++s2) s += p[(size_t)s2 * EE];
        float m = s * (1.0f / SS);
        hbar[b * EE + e]  = m;
        hbarb[b * EE + e] = (__bf16)m;
    }
}

// ---------------------------------------------------------------------------
// Register-blocked WMMA GEMM: C[M,N] = A[M,K] * B[K,N] (+bias, +residual, relu)
//   Each wave owns a (16*WM) x (16*WN) macro-tile.  Leading dims LDA/LDB are
//   compile-time so all fragment loads are base + 24-bit immediate offset
//   (one address pair per matrix, one uniform increment per K step).
//   Inner loop: WM A-frags resident, then per B-frag do WM WMMAs -> peak live
//   set is WM+1 fragments + WM*WN accumulators.  __launch_bounds__(256,1)
//   gives the compiler the full VGPR file (4x4: 128 acc + 40 frag regs).
//   A bf16 row-major; B bf16 row-major [K,N] or, if TB, stored [N,K].
//   Batched via gridDim.y.  Fragment layouts per CDNA5 ISA 7.12.2; tile index
//   wave-uniform so EXEC stays all-1s for WMMA.
// ---------------------------------------------------------------------------
template <int WM, int WN, int LDA, int LDB, bool TB, bool BIAS, bool DORELU,
          bool RES, bool OF, bool OB>
__global__ __launch_bounds__(256, 1)
void gemm_wmma_k(const __bf16* __restrict__ A, long long bsA,
                 const __bf16* __restrict__ Bm, long long bsB,
                 const float* __restrict__ bias,
                 const float* __restrict__ Rsd, int ldr, long long bsR,
                 float* __restrict__ Cf, __bf16* __restrict__ Cb,
                 int ldc, long long bsC,
                 int M, int N, int K) {
    const int tilesN = N / (16 * WN);
    const int ntiles = (M / (16 * WM)) * tilesN;
    const int tile   = blockIdx.x * (blockDim.x >> 5) + (threadIdx.x >> 5);
    if (tile >= ntiles) return;                 // wave-uniform
    const int m0   = (tile / tilesN) * (16 * WM);
    const int n0   = (tile % tilesN) * (16 * WN);
    const int lane = threadIdx.x & 31;
    const int half = lane >> 4;                 // 0: K 0-7/16-23, 1: K 8-15/24-31
    const int l16  = lane & 15;
    const long long b = blockIdx.y;

    // Per-lane base pointers; all per-fragment deltas are compile-time.
    const __bf16* aBase = A + b * bsA + (size_t)(m0 + l16) * LDA + half * 8;
    const __bf16* bBase;
    if constexpr (TB) bBase = Bm + b * bsB + (size_t)(n0 + l16) * LDB + half * 8;
    else              bBase = Bm + b * bsB + (size_t)(half * 8) * LDB + n0 + l16;

    v8f acc[WM][WN];
    #pragma unroll
    for (int mi = 0; mi < WM; ++mi)
        #pragma unroll
        for (int ni = 0; ni < WN; ++ni) acc[mi][ni] = {};

    union FragU { v16bf v; v8bf h[2]; };

    for (int kk = 0; kk < K; kk += 32) {
        // A fragments: lane l16 = row; two contiguous 8-elem runs (2x b128)
        FragU af[WM];
        #pragma unroll
        for (int mi = 0; mi < WM; ++mi) {
            af[mi].h[0] = *(const v8bf*)(aBase + mi * 16 * LDA);
            af[mi].h[1] = *(const v8bf*)(aBase + mi * 16 * LDA + 16);
        }
        // One B fragment at a time; fire its WM WMMAs immediately.
        #pragma unroll
        for (int ni = 0; ni < WN; ++ni) {
            FragU bf_;
            if constexpr (TB) {
                v16bf t = {};
                #pragma unroll
                for (int j = 0; j < 16; ++j) t[j] = bBase[ni * 16 * LDB + j + (j >> 3) * 8];
                bf_.v = t;
            } else {
                v16bf t = {};
                #pragma unroll
                for (int j = 0; j < 16; ++j) t[j] = bBase[ni * 16 + (j + (j >> 3) * 8) * LDB];
                bf_.v = t;
            }
            #pragma unroll
            for (int mi = 0; mi < WM; ++mi)
                acc[mi][ni] = __builtin_amdgcn_wmma_f32_16x16x32_bf16(
                    false, af[mi].v, false, bf_.v, (short)0, acc[mi][ni],
                    false, false);
        }
        aBase += 32;                                   // K advance (row-major A)
        if constexpr (TB) bBase += 32;                 // K along rows of [N,K]
        else              bBase += (size_t)32 * LDB;   // K along columns of [K,N]
    }

    #pragma unroll
    for (int ni = 0; ni < WN; ++ni) {
        const int n = n0 + 16 * ni + l16;
        float bval = 0.f;
        if constexpr (BIAS) bval = bias[n];
        #pragma unroll
        for (int mi = 0; mi < WM; ++mi) {
            #pragma unroll
            for (int r = 0; r < 8; ++r) {
                int m = m0 + 16 * mi + r + half * 8;   // C layout: VGPR r, lane half
                float v = acc[mi][ni][r] + bval;
                if constexpr (RES) v += Rsd[b * bsR + (size_t)m * ldr + n];
                if constexpr (DORELU) v = fmaxf(v, 0.0f);
                if constexpr (OF) Cf[b * bsC + (size_t)m * ldc + n] = v;
                if constexpr (OB) Cb[b * bsC + (size_t)m * ldc + n] = (__bf16)v;
            }
        }
    }
}

// ---------------------------------------------------------------------------
static inline unsigned cdiv(unsigned a, unsigned b) { return (a + b - 1) / b; }

extern "C" void kernel_launch(void* const* d_in, const int* in_sizes, int n_in,
                              void* d_out, int out_size, void* d_ws, size_t ws_size,
                              hipStream_t stream) {
    (void)in_sizes; (void)n_in; (void)out_size; (void)ws_size;
    const int*   x     = (const int*)d_in[0];
    const float* emb   = (const float*)d_in[1];
    const float* kqv_w = (const float*)d_in[2];   // (3, 512, 1536)
    const float* kqv_b = (const float*)d_in[3];   // (3, 1536)
    const float* g_w   = (const float*)d_in[4];   // (3, 512, 512)
    const float* g_b   = (const float*)d_in[5];   // (3, 512)
    const float* w0    = (const float*)d_in[6];   // (512, 1024)
    const float* b0    = (const float*)d_in[7];   // (1024,)
    const float* w1    = (const float*)d_in[8];   // (1024, 1024)
    const float* b1    = (const float*)d_in[9];   // (1024,)
    float* out = (float*)d_out;

    // -------- workspace carve-out (256B aligned) --------
    char* ws = (char*)d_ws;
    size_t off = 0;
    auto alloc = [&](size_t bytes) -> void* {
        void* p = ws + off;
        off = (off + bytes + 255) & ~(size_t)255;
        return p;
    };
    float*  h    = (float*) alloc((size_t)NTOK * EE * 4);          // 32 MB
    __bf16* hn   = (__bf16*)alloc((size_t)NTOK * EE * 2);          // 16 MB (also t=h+f)
    __bf16* y    = (__bf16*)alloc((size_t)NTOK * 3 * EE * 2);      // 48 MB
    float*  sc   = (float*) alloc((size_t)BB * SS * SS * 4);       // 32 MB
    __bf16* attn = (__bf16*)alloc((size_t)BB * SS * SS * 2);       // 16 MB
    __bf16* wq   = (__bf16*)alloc((size_t)NL * EE * 3 * EE * 2);
    __bf16* wg   = (__bf16*)alloc((size_t)NL * EE * EE * 2);
    __bf16* w0b  = (__bf16*)alloc((size_t)EE * 1024 * 2);
    __bf16* w1b  = (__bf16*)alloc((size_t)1024 * 1024 * 2);
    float*  mean = (float*) alloc(EE * 4);
    float*  rstd = (float*) alloc(EE * 4);
    float*  hbar = (float*) alloc(BB * EE * 4);
    __bf16* hbarb= (__bf16*)alloc(BB * EE * 2);
    __bf16* m0b  = (__bf16*)alloc(BB * 1024 * 2);

    // -------- convert weights to bf16 --------
    {
        int n;
        n = NL * EE * 3 * EE; f32_to_bf16_k<<<cdiv(n, 256), 256, 0, stream>>>(kqv_w, wq, n);
        n = NL * EE * EE;     f32_to_bf16_k<<<cdiv(n, 256), 256, 0, stream>>>(g_w, wg, n);
        n = EE * 1024;        f32_to_bf16_k<<<cdiv(n, 256), 256, 0, stream>>>(w0, w0b, n);
        n = 1024 * 1024;      f32_to_bf16_k<<<cdiv(n, 256), 256, 0, stream>>>(w1, w1b, n);
    }

    // -------- embedding + positional encoding --------
    embed_pos_k<<<NTOK, 256, 0, stream>>>(x, emb, h);

    const long long bsY = (long long)SS * 3 * EE;   // per-batch stride into y
    const long long bsS = (long long)SS * SS;       // per-batch stride scores/attn
    const long long bsH = (long long)SS * EE;       // per-batch stride into h

    for (int i = 0; i < NL; ++i) {
        // batchnorm over (B,S) per feature -> hn (bf16)
        bn_stats_k<<<EE, 256, 0, stream>>>(h, mean, rstd);
        bn_apply_k<<<cdiv(NTOK * EE, 256), 256, 0, stream>>>(h, mean, rstd, hn);

        // y = hn @ kqv_w[i] + kqv_b[i]   (16384 x 1536 x 512), bf16 out
        {
            unsigned ntiles = (NTOK / 64) * (3 * EE / 64);   // 6144 macro tiles
            gemm_wmma_k<4, 4, EE, 3 * EE, false, true, false, false, false, true>
                <<<dim3(cdiv(ntiles, 8), 1), 256, 0, stream>>>(
                    hn, 0, wq + (size_t)i * EE * 3 * EE, 0,
                    kqv_b + (size_t)i * 3 * EE,
                    nullptr, 0, 0, nullptr, y, 3 * EE, 0,
                    NTOK, 3 * EE, EE);
        }
        // scores[b] = q[b] @ k[b]^T   (512 x 512 x 512, batched over 32)
        {
            unsigned ntiles = (SS / 64) * (SS / 32);         // 128 macro tiles
            gemm_wmma_k<4, 2, 3 * EE, 3 * EE, true, false, false, false, true, false>
                <<<dim3(cdiv(ntiles, 8), BB), 256, 0, stream>>>(
                    y + EE, bsY,                // q slice
                    y, bsY,                     // k slice, transposed access
                    nullptr, nullptr, 0, 0,
                    sc, nullptr, SS, bsS,
                    SS, SS, EE);
        }
        // softmax rows -> attn (bf16)
        softmax_rows_k<<<cdiv(BB * SS, 8), 256, 0, stream>>>(sc, attn);

        // t = h + attn @ v   -> hn (bf16), residual fused in epilogue
        {
            unsigned ntiles = (SS / 64) * (EE / 64);         // 64 macro tiles
            gemm_wmma_k<4, 4, SS, 3 * EE, false, false, false, true, false, true>
                <<<dim3(cdiv(ntiles, 8), BB), 256, 0, stream>>>(
                    attn, bsS,
                    y + 2 * EE, bsY,            // v slice
                    nullptr,
                    h, EE, bsH,                 // residual
                    nullptr, hn, EE, bsH,
                    SS, EE, EE);
        }
        // h = relu(t @ g_w[i] + g_b[i])   (16384 x 512 x 512), f32 out
        {
            unsigned ntiles = (NTOK / 64) * (EE / 64);       // 2048 macro tiles
            gemm_wmma_k<4, 4, EE, EE, false, true, true, false, true, false>
                <<<dim3(cdiv(ntiles, 8), 1), 256, 0, stream>>>(
                    hn, 0, wg + (size_t)i * EE * EE, 0,
                    g_b + (size_t)i * EE,
                    nullptr, 0, 0, h, nullptr, EE, 0,
                    NTOK, EE, EE);
        }
    }

    // hbar = mean over s
    mean_seq_k<<<BB, 256, 0, stream>>>(h, hbar, hbarb);

    // mlp0: relu(hbar @ w0 + b0) -> bf16 (32 x 1024 x 512)
    {
        unsigned ntiles = (BB / 32) * (1024 / 64);           // 16 macro tiles
        gemm_wmma_k<2, 4, EE, 1024, false, true, true, false, false, true>
            <<<dim3(cdiv(ntiles, 8), 1), 256, 0, stream>>>(
                hbarb, 0, w0b, 0, b0,
                nullptr, 0, 0, nullptr, m0b, 1024, 0,
                BB, 1024, EE);
    }
    // mlp1: relu(m0 @ w1 + b1) -> d_out f32 (32 x 1024 x 1024)
    {
        unsigned ntiles = (BB / 32) * (1024 / 64);           // 16 macro tiles
        gemm_wmma_k<2, 4, 1024, 1024, false, true, true, false, true, false>
            <<<dim3(cdiv(ntiles, 8), 1), 256, 0, stream>>>(
                m0b, 0, w1b, 0, b1,
                nullptr, 0, 0, out, nullptr, 1024, 0,
                BB, 1024, 1024);
    }
}
